// MultiHeadAttention_33835752358084
// MI455X (gfx1250) — matrix operand order repaired and checked by reading, not compiled, not run on hardware
//
#include <hip/hip_runtime.h>

// ---------------------------------------------------------------------------
// MHA forward for MI455X (gfx1250): bf16 WMMA (f32 accum) + async LDS staging.
// B=4, S=1024, D=1024, H=16, DH=64
// ---------------------------------------------------------------------------

static constexpr int kB  = 4;
static constexpr int kS  = 1024;
static constexpr int kD  = 1024;
static constexpr int kH  = 16;
static constexpr int kDH = 64;

typedef __attribute__((ext_vector_type(16))) __bf16 v16bf;
typedef __attribute__((ext_vector_type(8)))  float  v8f;

union FragU {           // 32-byte bf16 fragment, assembled from two 16B loads
    uint4  q[2];
    v16bf  v;
};
union VF {              // f32 accumulator fragment with scalar access
    v8f   v;
    float f[8];
};

__device__ __forceinline__ v8f wmma_bf16(v16bf a, v16bf b, v8f c) {
    return __builtin_amdgcn_wmma_f32_16x16x32_bf16(
        false, a, false, b, (short)0, c, false, false);
}

// Async DMA: 16 bytes per lane, global -> LDS, tracked by ASYNCcnt.
//   lds_addr  : per-lane LDS byte address (16B aligned)
//   gvoff     : per-lane 32-bit byte offset added to scalar base (GVS mode)
__device__ __forceinline__ void async_b128(unsigned lds_addr, unsigned gvoff,
                                           const void* base) {
    asm volatile("global_load_async_to_lds_b128 %0, %1, %2"
                 :: "v"(lds_addr), "v"(gvoff), "s"(base) : "memory");
}

// A-fragment (16x32 bf16, row-major, stride in elements):
//   a[e] = A[m][(e/8)*16 + half*8 + (e%8)]
__device__ __forceinline__ v16bf load_a_frag(const __bf16* base, int stride,
                                             int mrow, int k0, int half) {
    const __bf16* p = base + (size_t)mrow * stride + k0 + half * 8;
    FragU f;
    f.q[0] = *(const uint4*)(p);
    f.q[1] = *(const uint4*)(p + 16);
    return f.v;
}

// B-fragment (32x16) from N-major storage Bt[n][k]:
//   b[e] = B[k0 + (e/8)*16 + half*8 + (e%8)][n]
__device__ __forceinline__ v16bf load_b_frag(const __bf16* base, int stride,
                                             int nrow, int k0, int half) {
    const __bf16* p = base + (size_t)nrow * stride + k0 + half * 8;
    FragU f;
    f.q[0] = *(const uint4*)(p);
    f.q[1] = *(const uint4*)(p + 16);
    return f.v;
}

// ---------------------------------------------------------------------------
// Preprocessing 1: x_{Q,K,V} fp32 -> bf16, contiguous [sel][B][S][D]
// ---------------------------------------------------------------------------
__global__ __launch_bounds__(256) void cvt_x_kernel(
    const float* __restrict__ xQ, const float* __restrict__ xK,
    const float* __restrict__ xV, __bf16* __restrict__ Xb)
{
    const size_t per = (size_t)kB * kS * kD;                // 4M elements
    size_t idx = ((size_t)blockIdx.x * 256 + threadIdx.x) * 8;
    int    sel = (int)(idx / per);
    const float* X = (sel == 0) ? xQ : (sel == 1) ? xK : xV;
    size_t off = idx - (size_t)sel * per;
    float4 f0 = *(const float4*)(X + off);
    float4 f1 = *(const float4*)(X + off + 4);
    union { __bf16 h[8]; uint4 q; } u;
    u.h[0] = (__bf16)f0.x; u.h[1] = (__bf16)f0.y;
    u.h[2] = (__bf16)f0.z; u.h[3] = (__bf16)f0.w;
    u.h[4] = (__bf16)f1.x; u.h[5] = (__bf16)f1.y;
    u.h[6] = (__bf16)f1.z; u.h[7] = (__bf16)f1.w;
    *(uint4*)(Xb + idx) = u.q;
}

// ---------------------------------------------------------------------------
// Preprocessing 2: weights -> bf16, transposed to N-major.
//   Wt  [sel][h][n][k] = W_sel[h][k][n]     (3*16*64*1024 elements)
//   Wot [n][k]         = Wo[k][n]           (1024*1024 elements)
// ---------------------------------------------------------------------------
__global__ __launch_bounds__(256) void cvt_w_kernel(
    const float* __restrict__ Wq, const float* __restrict__ Wk,
    const float* __restrict__ Wv, const float* __restrict__ Wo,
    __bf16* __restrict__ Wt, __bf16* __restrict__ Wot)
{
    const size_t wqkv = (size_t)3 * kH * kD * kDH;          // 3M
    size_t i = (size_t)blockIdx.x * 256 + threadIdx.x;
    if (i < wqkv) {
        int    sel = (int)(i / ((size_t)kH * kD * kDH));
        size_t r   = i % ((size_t)kH * kD * kDH);
        int h = (int)(r / (kD * kDH));
        size_t r2 = r % (kD * kDH);
        int n = (int)(r2 / kD);
        int k = (int)(r2 % kD);
        const float* W = (sel == 0) ? Wq : (sel == 1) ? Wk : Wv;
        Wt[i] = (__bf16)W[(size_t)h * kD * kDH + (size_t)k * kDH + n];
    } else {
        size_t j = i - wqkv;                                 // 1M
        int n = (int)(j / kD), k = (int)(j % kD);
        Wot[j] = (__bf16)Wo[(size_t)k * kD + n];
    }
}

// ---------------------------------------------------------------------------
// Kernel 1: per-head QKV projection.  grid = (S/128, B*H, 3)
//   A tiles (128x32) and B tiles (64x32, pre-transposed W) staged with
//   double-buffered global_load_async_to_lds_b128 (3 issues / tile / wave).
// ---------------------------------------------------------------------------
__global__ __launch_bounds__(256) void qkv_proj_kernel(
    const __bf16* __restrict__ Xb, const __bf16* __restrict__ Wt,
    const float* __restrict__ bq, const float* __restrict__ bk,
    const float* __restrict__ bv,
    __bf16* __restrict__ Qb, __bf16* __restrict__ Kb, __bf16* __restrict__ Vt)
{
    const int sel = blockIdx.z;
    const int bh  = blockIdx.y;
    const int b   = bh / kH, h = bh % kH;
    const int m0  = blockIdx.x * 128;

    const __bf16* X    = Xb + ((size_t)sel * kB + b) * kS * kD;       // [s][d]
    const __bf16* W    = Wt + ((size_t)sel * kH + h) * kDH * kD;      // [n][k]
    const float*  bias = ((sel == 0) ? bq : (sel == 1) ? bk : bv) + h * kDH;

    // LDS: A0 A1 (128x40 each) | B0 B1 (64x40 each)
    __shared__ alignas(16) __bf16 smem[2 * 128 * 40 + 2 * 64 * 40];
    const unsigned lds0 = (unsigned)(uintptr_t)(void*)smem;
    const int ASZ = 128 * 40, BOFF = 2 * ASZ, BSZ = 64 * 40;

    const int tid  = threadIdx.x;
    const int lane = tid & 31, wid = tid >> 5;
    const int lrow = lane & 15, half = lane >> 4;

    auto stage = [&](int kt, int buf) {
        const int k0 = kt * 32;
        unsigned ab = lds0 + (unsigned)(buf * ASZ) * 2;
        for (int i = 0; i < 2; ++i) {                       // A: 512 chunks
            int cid = i * 256 + tid;
            int row = cid >> 2, c = cid & 3;
            async_b128(ab + row * 80 + c * 16,
                       (unsigned)(((m0 + row) * kD + k0 + c * 8) * 2), X);
        }
        unsigned bb = lds0 + (unsigned)(BOFF + buf * BSZ) * 2;
        {                                                   // B: 256 chunks
            int row = tid >> 2, c = tid & 3;
            async_b128(bb + row * 80 + c * 16,
                       (unsigned)((row * kD + k0 + c * 8) * 2), W);
        }
    };

    VF acc[4];
    for (int t = 0; t < 4; ++t)
        for (int r = 0; r < 8; ++r) acc[t].f[r] = 0.0f;

    const int NT = kD / 32;
    stage(0, 0);
    for (int kt = 0; kt < NT; ++kt) {
        if (kt + 1 < NT) {
            stage(kt + 1, (kt + 1) & 1);
            asm volatile("s_wait_asynccnt 3" ::: "memory");
        } else {
            asm volatile("s_wait_asynccnt 0" ::: "memory");
        }
        __syncthreads();

        const __bf16* As = smem + (kt & 1) * ASZ;
        const __bf16* Bs = smem + BOFF + (kt & 1) * BSZ;
        v16bf a = load_a_frag(As, 40, wid * 16 + lrow, 0, half);
        for (int t = 0; t < 4; ++t) {
            v16bf bm = load_b_frag(Bs, 40, t * 16 + lrow, 0, half);
            acc[t].v = wmma_bf16(a, bm, acc[t].v);
        }
        __syncthreads();
    }

    const int srow0 = m0 + wid * 16;
    for (int t = 0; t < 4; ++t) {
        const int   dh = t * 16 + lrow;
        const float bv2 = bias[dh];
        for (int r = 0; r < 8; ++r) {
            float v    = acc[t].f[r] + bv2;
            int   srow = srow0 + r + 8 * half;
            if (sel < 2) {
                __bf16* out = (sel == 0) ? Qb : Kb;
                out[(((size_t)b * kH + h) * kS + srow) * kDH + dh] = (__bf16)v;
            } else {
                Vt[(((size_t)b * kH + h) * kDH + dh) * kS + srow] = (__bf16)v;
            }
        }
    }
}

// ---------------------------------------------------------------------------
// Kernel 2: attention, online softmax.  grid = (S/128, B*H)
//   64-key K/V tiles staged cooperatively via async DMA (4 issues/tile),
//   double-buffered; probabilities spilled per-wave to LDS for the AV WMMA.
// ---------------------------------------------------------------------------
__global__ __launch_bounds__(256) void attn_kernel(
    const __bf16* __restrict__ Qb, const __bf16* __restrict__ Kb,
    const __bf16* __restrict__ Vt, const int* __restrict__ lens,
    __bf16* __restrict__ AVc)
{
    const int bh = blockIdx.y;
    const int b  = bh / kH, h = bh % kH;
    const int m0 = blockIdx.x * 128;

    const int tid  = threadIdx.x;
    const int lane = tid & 31, wid = tid >> 5;
    const int lrow = lane & 15, half = lane >> 4;

    const size_t bhoff = (size_t)b * kH + h;
    const __bf16* Q = Qb + bhoff * kS * kDH;   // [s][dh]
    const __bf16* K = Kb + bhoff * kS * kDH;   // [t][dh]
    const __bf16* V = Vt + bhoff * kDH * kS;   // [dh][t]
    const int len   = lens[b];

    // LDS: K0 K1 V0 V1 (64x72 each, stride 72 vs bank conflicts) | P (8x16x40)
    __shared__ alignas(16) __bf16 smem[4 * 64 * 72 + 8 * 16 * 40];
    const unsigned lds0 = (unsigned)(uintptr_t)(void*)smem;
    const int KSZ = 64 * 72;
    __bf16* Pw = smem + 4 * KSZ + wid * (16 * 40);

    auto stage_kv = [&](int key0, int buf) {
        unsigned kb = lds0 + (unsigned)(buf * KSZ) * 2;
        unsigned vb = lds0 + (unsigned)((2 + buf) * KSZ) * 2;
        for (int i = 0; i < 2; ++i) {                       // K: 512 chunks
            int cid = i * 256 + tid;
            int row = cid >> 3, c = cid & 7;
            async_b128(kb + row * 144 + c * 16,
                       (unsigned)(((key0 + row) * kDH + c * 8) * 2), K);
        }
        for (int i = 0; i < 2; ++i) {                       // V: 512 chunks
            int cid = i * 256 + tid;
            int row = cid >> 3, c = cid & 7;
            async_b128(vb + row * 144 + c * 16,
                       (unsigned)((row * kS + key0 + c * 8) * 2), V);
        }
    };

    const int mrow = m0 + wid * 16 + lrow;
    v16bf qa0 = load_a_frag(Q, kDH, mrow, 0, half);
    v16bf qa1 = load_a_frag(Q, kDH, mrow, 32, half);

    float rmax[8], rsum[8];
    bool  rowvalid[8];
    VF    acc[4];
    for (int r = 0; r < 8; ++r) {
        rmax[r] = -1e30f;
        rsum[r] = 0.0f;
        rowvalid[r] = (m0 + wid * 16 + r + 8 * half) < len;
    }
    for (int t = 0; t < 4; ++t)
        for (int r = 0; r < 8; ++r) acc[t].f[r] = 0.0f;

    const float scale = 0.125f;   // 1/sqrt(64)
    const int   NT    = kS / 64;

    stage_kv(0, 0);
    for (int kt = 0; kt < NT; ++kt) {
        if (kt + 1 < NT) {
            stage_kv((kt + 1) * 64, (kt + 1) & 1);
            asm volatile("s_wait_asynccnt 4" ::: "memory");
        } else {
            asm volatile("s_wait_asynccnt 0" ::: "memory");
        }
        __syncthreads();

        const __bf16* Ks = smem + (kt & 1) * KSZ;
        const __bf16* Vs = smem + (2 + (kt & 1)) * KSZ;

        for (int sub = 0; sub < 2; ++sub) {
            const int kl0 = sub * 32;
            // scores: 16 x 32 keys, K-dim = dh = 64
            VF sc[2];
            for (int nt = 0; nt < 2; ++nt) {
                for (int r = 0; r < 8; ++r) sc[nt].f[r] = 0.0f;
                v16bf kb0 = load_b_frag(Ks, 72, kl0 + nt * 16 + lrow, 0, half);
                v16bf kb1 = load_b_frag(Ks, 72, kl0 + nt * 16 + lrow, 32, half);
                sc[nt].v = wmma_bf16(qa0, kb0, sc[nt].v);
                sc[nt].v = wmma_bf16(qa1, kb1, sc[nt].v);
            }
            // scale + row mask + online softmax (row m = r + 8*half)
            for (int r = 0; r < 8; ++r) {
                float x0 = rowvalid[r] ? sc[0].f[r] * scale : 0.0f;
                float x1 = rowvalid[r] ? sc[1].f[r] * scale : 0.0f;
                float mx = fmaxf(x0, x1);
                for (int m = 1; m < 16; m <<= 1)
                    mx = fmaxf(mx, __shfl_xor(mx, m, 32));
                float nm   = fmaxf(rmax[r], mx);
                float corr = __expf(rmax[r] - nm);
                float e0   = __expf(x0 - nm);
                float e1   = __expf(x1 - nm);
                float ts   = e0 + e1;
                for (int m = 1; m < 16; m <<= 1)
                    ts += __shfl_xor(ts, m, 32);
                rsum[r] = rsum[r] * corr + ts;
                rmax[r] = nm;
                for (int t = 0; t < 4; ++t) acc[t].f[r] *= corr;
                Pw[(r + 8 * half) * 40 + lrow]      = (__bf16)e0;
                Pw[(r + 8 * half) * 40 + 16 + lrow] = (__bf16)e1;
            }
            asm volatile("s_wait_dscnt 0" ::: "memory");   // intra-wave RAW

            v16bf pa = load_a_frag(Pw, 40, lrow, 0, half);
            for (int t = 0; t < 4; ++t) {
                v16bf vb = load_b_frag(Vs, 72, t * 16 + lrow, kl0, half);
                acc[t].v = wmma_bf16(pa, vb, acc[t].v);
            }
        }
        __syncthreads();
    }

    for (int t = 0; t < 4; ++t) {
        for (int r = 0; r < 8; ++r) {
            float v    = acc[t].f[r] / rsum[r];
            int   srow = m0 + wid * 16 + r + 8 * half;
            int   col  = h * kDH + t * 16 + lrow;
            AVc[((size_t)b * kS + srow) * kD + col] = (__bf16)v;
        }
    }
}

// ---------------------------------------------------------------------------
// Kernel 3: output projection.  grid = (B*S/128, D/64)
//   A fragments direct from contiguous bf16 context; B tile async-staged
//   from pre-transposed Wo (1 issue / tile / wave, double-buffered).
// ---------------------------------------------------------------------------
__global__ __launch_bounds__(256) void out_proj_kernel(
    const __bf16* __restrict__ AVc, const __bf16* __restrict__ Wot,
    const float* __restrict__ bo, float* __restrict__ out)
{
    const int m0 = blockIdx.x * 128;
    const int n0 = blockIdx.y * 64;

    const int tid  = threadIdx.x;
    const int lane = tid & 31, wid = tid >> 5;
    const int lrow = lane & 15, half = lane >> 4;

    __shared__ alignas(16) __bf16 smem[2 * 64 * 40];
    const unsigned lds0 = (unsigned)(uintptr_t)(void*)smem;
    const int BSZ = 64 * 40;

    auto stage_b = [&](int kt, int buf) {
        const int k0 = kt * 32;
        unsigned bb = lds0 + (unsigned)(buf * BSZ) * 2;
        int row = tid >> 2, c = tid & 3;
        async_b128(bb + row * 80 + c * 16,
                   (unsigned)(((n0 + row) * kD + k0 + c * 8) * 2), Wot);
    };

    VF acc[4];
    for (int t = 0; t < 4; ++t)
        for (int r = 0; r < 8; ++r) acc[t].f[r] = 0.0f;

    const int mrow = m0 + wid * 16 + lrow;
    const int NT   = kD / 32;

    stage_b(0, 0);
    for (int kt = 0; kt < NT; ++kt) {
        if (kt + 1 < NT) {
            stage_b(kt + 1, (kt + 1) & 1);
            asm volatile("s_wait_asynccnt 1" ::: "memory");
        } else {
            asm volatile("s_wait_asynccnt 0" ::: "memory");
        }
        __syncthreads();

        const __bf16* Bs = smem + (kt & 1) * BSZ;
        v16bf a = load_a_frag(AVc, kD, mrow, kt * 32, half);
        for (int t = 0; t < 4; ++t) {
            v16bf bm = load_b_frag(Bs, 40, t * 16 + lrow, 0, half);
            acc[t].v = wmma_bf16(a, bm, acc[t].v);
        }
        __syncthreads();
    }

    for (int t = 0; t < 4; ++t) {
        const int   n  = n0 + t * 16 + lrow;
        const float bv = bo[n];
        for (int r = 0; r < 8; ++r) {
            int row = m0 + wid * 16 + r + 8 * half;
            out[(size_t)row * kD + n] = acc[t].f[r] + bv;
        }
    }
}

// ---------------------------------------------------------------------------
// Launch
// ---------------------------------------------------------------------------
extern "C" void kernel_launch(void* const* d_in, const int* in_sizes, int n_in,
                              void* d_out, int out_size, void* d_ws, size_t ws_size,
                              hipStream_t stream)
{
    const float* xQ   = (const float*)d_in[0];
    const float* xK   = (const float*)d_in[1];
    const float* xV   = (const float*)d_in[2];
    const int*   lens = (const int*)d_in[3];
    const float* Wq   = (const float*)d_in[4];
    const float* bq   = (const float*)d_in[5];
    const float* Wk   = (const float*)d_in[6];
    const float* bk   = (const float*)d_in[7];
    const float* Wv   = (const float*)d_in[8];
    const float* bv   = (const float*)d_in[9];
    const float* Wo   = (const float*)d_in[10];
    const float* bo   = (const float*)d_in[11];
    float*       out  = (float*)d_out;

    const size_t perQKV = (size_t)kB * kH * kS * kDH;       // 4M elements
    const size_t perX   = (size_t)kB * kS * kD;             // 4M elements
    __bf16* Qb  = (__bf16*)d_ws;                            //  8 MiB
    __bf16* Kb  = Qb + perQKV;                              //  8 MiB
    __bf16* Vt  = Kb + perQKV;                              //  8 MiB [B,H,DH,S]
    __bf16* AVc = Vt + perQKV;                              //  8 MiB [B,S,H*DH]
    __bf16* Xb  = AVc + perX;                               // 24 MiB [3,B,S,D]
    __bf16* Wt  = Xb + 3 * perX;                            //  6 MiB [3,H,DH,D]
    __bf16* Wot = Wt + (size_t)3 * kH * kDH * kD;           //  2 MiB [D,D]

    dim3 blk(256);

    cvt_x_kernel<<<dim3((unsigned)(3 * perX / (256 * 8))), blk, 0, stream>>>(
        xQ, xK, xV, Xb);

    cvt_w_kernel<<<dim3((unsigned)((3 * (size_t)kH * kD * kDH +
                                    (size_t)kD * kD) / 256)), blk, 0, stream>>>(
        Wq, Wk, Wv, Wo, Wt, Wot);

    qkv_proj_kernel<<<dim3(kS / 128, kB * kH, 3), blk, 0, stream>>>(
        Xb, Wt, bq, bk, bv, Qb, Kb, Vt);

    attn_kernel<<<dim3(kS / 128, kB * kH), blk, 0, stream>>>(
        Qb, Kb, Vt, lens, AVc);

    out_proj_kernel<<<dim3((kB * kS) / 128, kD / 64), blk, 0, stream>>>(
        AVc, Wot, bo, out);
}
